// ScaledDotProductAttention_37898791420398
// MI455X (gfx1250) — matrix operand order repaired and checked by reading, not compiled, not run on hardware
//
#include <hip/hip_runtime.h>
#include <stdint.h>

// ---------------------------------------------------------------------------
// Flash attention for B=2, H=16, S=2048, D=64 (fp32 in/out, bf16 WMMA MACs).
// Block = 4 waves = 128 threads; each wave owns 16 query rows (block = 64).
// Grid = B*H*(S/64) = 1024 blocks. K/V streamed in 64-key tiles through LDS
// (float4 global loads, fp32->bf16 on the way in, V transposed for PV frags).
// Online softmax state (m, l) kept per-lane; cross-lane reductions via
// __shfl_xor over the 16-lane half that shares each C-matrix row.
// ---------------------------------------------------------------------------

typedef __bf16 v16bf __attribute__((ext_vector_type(16)));
typedef __bf16 v8bf  __attribute__((ext_vector_type(8)));
typedef __bf16 v4bf  __attribute__((ext_vector_type(4)));
typedef float  v8f   __attribute__((ext_vector_type(8)));

#define WMMA_BF16(a, b, c) \
  __builtin_amdgcn_wmma_f32_16x16x32_bf16(false, (a), false, (b), (short)0, (c), false, false)

// float -> bf16: use the native conversion (single v_cvt on gfx1250).
__device__ __forceinline__ __bf16 f2bf(float f) {
#if defined(__gfx1250__)
  return (__bf16)f;
#else
  unsigned u = __builtin_bit_cast(unsigned, f);
  u += 0x7FFFu + ((u >> 16) & 1u);
  unsigned short h = (unsigned short)(u >> 16);
  return __builtin_bit_cast(__bf16, h);
#endif
}

__global__ __launch_bounds__(128)
void fa_fwd_kernel(const float* __restrict__ Q, const float* __restrict__ K,
                   const float* __restrict__ V, const uint8_t* __restrict__ mask,
                   float* __restrict__ O) {
  constexpr int S = 2048, D = 64, H = 16;
  constexpr int QT = 64;            // query rows per block
  constexpr int KT = 64;            // keys per main-loop step (4 x 16-key tiles)
  constexpr float scale = 0.125f;   // 1/sqrt(64)
  constexpr float NEG = -1.0e30f;

  __shared__ __align__(32) __bf16 Kl[KT][D];        // [key][d]      8 KB
  __shared__ __align__(32) __bf16 Vt[D][KT];        // [d][key]      8 KB
  __shared__ __align__(32) __bf16 Pl[4][16][KT];    // per-wave P    8 KB

  const int tid  = threadIdx.x;
  const int wave = tid >> 5;
  const int lane = tid & 31;
  const int half = (lane >> 4) & 1;   // 0: lanes 0-15, 1: lanes 16-31
  const int ln   = lane & 15;

  const int nqt = S / QT;
  const int bh  = blockIdx.x / nqt;   // fused (batch, head)
  const int qt  = blockIdx.x % nqt;
  const int b   = bh / H;

  const long base = (long)bh * S * D;
  const int  qrow = qt * QT + wave * 16 + ln;   // A-matrix row for this lane

  // ---- Q fragments (scale folded into the bf16 conversion) ----
  // A operand (16x32) in this code's contiguous k order: lane -> M = lane%16; element e -> K = e + 16*half
  v16bf aq0, aq1;
  #pragma unroll
  for (int e = 0; e < 16; ++e) {
    int kk = e + half * 16;
    aq0[e] = f2bf(Q[base + (long)qrow * D + kk] * scale);
    aq1[e] = f2bf(Q[base + (long)qrow * D + 32 + kk] * scale);
  }

  // ---- running state. C/D layout: VGPR v -> row M = v + 8*half, col N = lane%16.
  v8f acc0 = {}, acc1 = {}, acc2 = {}, acc3 = {};
  float mrow[8], lrow[8];
  #pragma unroll
  for (int v = 0; v < 8; ++v) { mrow[v] = NEG; lrow[v] = 0.0f; }

  for (int k0 = 0; k0 < S; k0 += KT) {
    // ---- cooperative K/V tile load: float4 global reads, bf16 LDS stores ----
    // 64x64 fp32 elements per matrix = 1024 float4 = 8 float4 per thread.
    #pragma unroll
    for (int i = 0; i < (KT * D) / (128 * 4); ++i) {
      int idx = tid + i * 128;            // float4 index; consecutive tids coalesce
      int r = idx >> 4, c4 = (idx & 15) * 4;
      const float4 kv = *(const float4*)&K[base + (long)(k0 + r) * D + c4];
      const float4 vv = *(const float4*)&V[base + (long)(k0 + r) * D + c4];
      v4bf kb; kb[0] = f2bf(kv.x); kb[1] = f2bf(kv.y); kb[2] = f2bf(kv.z); kb[3] = f2bf(kv.w);
      *(v4bf*)&Kl[r][c4] = kb;
      Vt[c4 + 0][r] = f2bf(vv.x);
      Vt[c4 + 1][r] = f2bf(vv.y);
      Vt[c4 + 2][r] = f2bf(vv.z);
      Vt[c4 + 3][r] = f2bf(vv.w);
    }
    // prefetch next 16KB K/V tiles (one 128B line per thread) -> global_prefetch_b8
    if (k0 + KT < S) {
      __builtin_prefetch(&K[base + (long)(k0 + KT) * D + tid * 32], 0, 1);
      __builtin_prefetch(&V[base + (long)(k0 + KT) * D + tid * 32], 0, 1);
    }
    __syncthreads();

    // ---- S = Q*K^T : four 16-key tiles, contraction over D = 64 = 2 x 32 ----
    // B operand (32x16) in this code's contiguous k order: lane -> N = lane%16; element e -> K = e + 16*half
    v8f st[4];
    #pragma unroll
    for (int t = 0; t < 4; ++t) {
      const __bf16* kp = &Kl[16 * t + ln][half * 16];
      v16bf bk0 = *(const v16bf*)kp;          // d in [0,32)
      v16bf bk1 = *(const v16bf*)(kp + 32);   // d in [32,64)
      v8f s = {};
      s = WMMA_BF16(aq0, bk0, s);
      s = WMMA_BF16(aq1, bk1, s);
      st[t] = s;
    }

    // ---- key mask: this lane holds key column k0 + 16t + ln of tile t ----
    bool mk[4];
    #pragma unroll
    for (int t = 0; t < 4; ++t) mk[t] = mask[(long)b * S + k0 + 16 * t + ln] != 0;

    // ---- online softmax update over the 64 new keys (one m/l update) ----
    #pragma unroll
    for (int v = 0; v < 8; ++v) {
      float x[4];
      #pragma unroll
      for (int t = 0; t < 4; ++t) x[t] = mk[t] ? st[t][v] : NEG;
      float mx = fmaxf(fmaxf(x[0], x[1]), fmaxf(x[2], x[3]));
      #pragma unroll
      for (int d = 1; d < 16; d <<= 1) mx = fmaxf(mx, __shfl_xor(mx, d, 32));
      float mnew = fmaxf(mrow[v], mx);
      float p[4], ps = 0.0f;
      #pragma unroll
      for (int t = 0; t < 4; ++t) { p[t] = __expf(x[t] - mnew); ps += p[t]; }
      #pragma unroll
      for (int d = 1; d < 16; d <<= 1) ps += __shfl_xor(ps, d, 32);
      float alpha = __expf(mrow[v] - mnew);
      mrow[v] = mnew;
      lrow[v] = lrow[v] * alpha + ps;
      acc0[v] *= alpha; acc1[v] *= alpha; acc2[v] *= alpha; acc3[v] *= alpha;
      // restage P: C-layout (row v+8*half, col 16t+ln) -> row-major LDS
      int r = v + half * 8;
      #pragma unroll
      for (int t = 0; t < 4; ++t) Pl[wave][r][16 * t + ln] = f2bf(p[t]);
    }

    // ---- P fragments: two 16x32 A-matrices (keys [0,32) and [32,64)) ----
    v16bf pa0, pa1;
    {
      const __bf16* pp = &Pl[wave][ln][half * 16];
      v8bf l0 = *(const v8bf*)pp;           // K = half*16 + 0..7
      v8bf h0 = *(const v8bf*)(pp + 8);     // K = half*16 + 8..15
      v8bf l1 = *(const v8bf*)(pp + 32);
      v8bf h1 = *(const v8bf*)(pp + 40);
      pa0 = __builtin_shufflevector(l0, h0, 0, 1, 2, 3, 4, 5, 6, 7,
                                            8, 9, 10, 11, 12, 13, 14, 15);
      pa1 = __builtin_shufflevector(l1, h1, 0, 1, 2, 3, 4, 5, 6, 7,
                                            8, 9, 10, 11, 12, 13, 14, 15);
    }

    // ---- O += P(16x64) * V(64x64): 2 K-chunks x 4 d-chunks = 8 WMMAs ----
    #pragma unroll
    for (int c = 0; c < 2; ++c) {
      v16bf pa = c ? pa1 : pa0;
      v16bf bv0 = *(const v16bf*)(&Vt[ 0 + ln][32 * c + half * 16]);
      v16bf bv1 = *(const v16bf*)(&Vt[16 + ln][32 * c + half * 16]);
      v16bf bv2 = *(const v16bf*)(&Vt[32 + ln][32 * c + half * 16]);
      v16bf bv3 = *(const v16bf*)(&Vt[48 + ln][32 * c + half * 16]);
      acc0 = WMMA_BF16(pa, bv0, acc0);
      acc1 = WMMA_BF16(pa, bv1, acc1);
      acc2 = WMMA_BF16(pa, bv2, acc2);
      acc3 = WMMA_BF16(pa, bv3, acc3);
    }

    __syncthreads();  // protect K/V tile before next cooperative load
  }

  // ---- epilogue: normalize by row sums and store ----
  #pragma unroll
  for (int v = 0; v < 8; ++v) {
    float inv = 1.0f / lrow[v];
    int q = qt * QT + wave * 16 + v + half * 8;
    long o = base + (long)q * D + ln;
    O[o]      = acc0[v] * inv;
    O[o + 16] = acc1[v] * inv;
    O[o + 32] = acc2[v] * inv;
    O[o + 48] = acc3[v] * inv;
  }
}

extern "C" void kernel_launch(void* const* d_in, const int* in_sizes, int n_in,
                              void* d_out, int out_size, void* d_ws, size_t ws_size,
                              hipStream_t stream) {
  (void)in_sizes; (void)n_in; (void)out_size; (void)d_ws; (void)ws_size;
  const float*   Q = (const float*)d_in[0];
  const float*   K = (const float*)d_in[1];
  const float*   V = (const float*)d_in[2];
  const uint8_t* M = (const uint8_t*)d_in[3];  // jnp.bool_ -> 1 byte per element
  float* O = (float*)d_out;

  constexpr int B = 2, H = 16, S = 2048, QT = 64;
  dim3 grid(B * H * (S / QT));
  fa_fwd_kernel<<<grid, 128, 0, stream>>>(Q, K, V, M, O);
}
